// ExternalMemory_42107859370130
// MI455X (gfx1250) — compile-verified, matrix operand order checked by reading, not run on hardware
//
#include <hip/hip_runtime.h>
#include <math.h>

#define MEM_SIZE 100000
#define EMB      1024
#define HEADS    8
#define HEAD_DIM 128
#define TOPK     10
#define BATCH    2048
#define NSPLIT   20
#define SPLIT_LEN (MEM_SIZE / NSPLIT)      /* 5000 */
#define BM       64                        /* block M tile (batch rows)   */
#define BN       64                        /* block N tile (memory rows)  */
#define KCHUNK   64                        /* K elems staged per B chunk  */
#define NCHUNKS  (EMB / KCHUNK)            /* 16 chunks per N tile        */
#define QLD      (EMB + 8)                 /* padded Q row stride (elems) */
#define BLD      (KCHUNK + 8)              /* padded B row stride (elems) */
#define NEG_INF  (-3.0e38f)

typedef __attribute__((ext_vector_type(16))) __bf16 v16bf;
typedef __attribute__((ext_vector_type(8)))  __bf16 v8bf;
typedef __attribute__((ext_vector_type(8)))  float  v8f;

// ---------- helpers ----------

__device__ __forceinline__ __bf16 f2bf(float f) {
    union { float f; unsigned u; } v; v.f = f;
    unsigned r = v.u + 0x7FFFu + ((v.u >> 16) & 1u);   // round-nearest-even
    unsigned short h = (unsigned short)(r >> 16);
    __bf16 b; __builtin_memcpy(&b, &h, 2); return b;
}
__device__ __forceinline__ float bf2f(__bf16 b) {
    unsigned short h; __builtin_memcpy(&h, &b, 2);
    union { unsigned u; float f; } v; v.u = ((unsigned)h) << 16; return v.f;
}

// 16-bit WMMA operand slice for this lane (row-major source).
// CDNA5 16-bit A/B layout: lane holds K = {0..7, 16..23} (lanes 0-15) or
// K = {8..15, 24..31} (lanes 16-31) -> two contiguous 16-byte chunks.
__device__ __forceinline__ v16bf ld_op16(const __bf16* __restrict__ p) {
    v8bf lo = *(const v8bf*)(p);
    v8bf hi = *(const v8bf*)(p + 16);
    v16bf o;
#pragma unroll
    for (int i = 0; i < 8; ++i) { o[i] = lo[i]; o[i + 8] = hi[i]; }
    return o;
}

__device__ __forceinline__ void topk_insert(float (&tv)[TOPK], int (&ti)[TOPK],
                                            float v, int id) {
    if (v <= tv[TOPK - 1]) return;           // cheap reject vs current 10th best
#pragma unroll
    for (int s = 0; s < TOPK; ++s) {
        if (v > tv[s]) {
            float fv = tv[s]; int fi = ti[s];
            tv[s] = v; ti[s] = id;
            v = fv; id = fi;
        }
    }
}

// ---------- fp32 -> bf16 conversion ----------

__global__ void cvt_f32_bf16(const float* __restrict__ src,
                             __bf16* __restrict__ dst, long n) {
    long i = (long)blockIdx.x * blockDim.x + threadIdx.x;
    long stride = (long)gridDim.x * blockDim.x;
    for (; i < n; i += stride) dst[i] = f2bf(src[i]);
}

// ---------- fused similarity GEMM + streaming top-k ----------
// Block = 256 threads (8 waves) owning a 64x64 (M x N) tile step:
// 4 M-groups x 2 N-groups, each wave a 16x32 strip (acc[2], A reused 2x).
// Q tile (64x1024 bf16) async-copied to LDS once. B streamed through a
// double-buffered LDS stage (64 rows x 64 K, 9KB) with
// global_load_async_to_lds_b128 + s_wait_asynccnt, pipelined across a flat
// (tile, k-chunk) sequence -> B global traffic cut 4x vs per-wave fetch.

__global__ __launch_bounds__(256)
void sim_topk_kernel(const __bf16* __restrict__ qBf,      // [B, E] bf16
                     const __bf16* __restrict__ memBf,    // [M, E] bf16
                     float* __restrict__ pVal,            // [B, NSPLIT, TOPK]
                     int*   __restrict__ pIdx) {
    __shared__ __bf16 smemQ[BM * QLD];                    // 132,096 B
    __shared__ __bf16 smemB[2][BN * BLD];                 // 18,432 B
    __shared__ float  smemS[8][256];                      // 8,192 B

    const int tid  = threadIdx.x;
    const int wave = tid >> 5;
    const int lane = tid & 31;
    const int mg   = wave & 3;                // M group (16 rows)
    const int ng   = wave >> 2;               // N group (32 cols)
    const int half = lane >> 4;
    const int q16  = lane & 15;

    const int mBase = blockIdx.x * BM;
    const int split = blockIdx.y;
    const int splitBase = split * SPLIT_LEN;
    const int limit = splitBase + SPLIT_LEN;
    const int numTiles = (SPLIT_LEN + BN - 1) / BN;       // 79
    const int total = numTiles * NCHUNKS;

    // ---- async copy of the 64x1024 bf16 Q tile into LDS ----
    for (int i = tid; i < BM * (EMB / 8); i += 256) {     // 16B chunks
        int r = i >> 7, c = (i & 127) << 3;
        unsigned lds = (unsigned)(size_t)&smemQ[r * QLD + c];
        const __bf16* gp = qBf + (size_t)(mBase + r) * EMB + c;
        asm volatile("global_load_async_to_lds_b128 %0, %1, off"
                     :: "v"(lds), "v"(gp) : "memory");
    }

    // ---- B chunk fill: 64 rows x 64 K elems -> 2 async b128 per thread ----
    auto issue_fill = [&](int g) {
        const int t  = g >> 4;                // NCHUNKS == 16
        const int kc = (g & 15) * KCHUNK;
        const int n0 = splitBase + t * BN;
        __bf16* dst = &smemB[g & 1][0];
#pragma unroll
        for (int i = tid; i < BN * (KCHUNK / 8); i += 256) {   // 512
            int r = i >> 3, c8 = (i & 7) << 3;
            int gr = n0 + r;
            if (gr > MEM_SIZE - 1) gr = MEM_SIZE - 1;          // clamp; masked
            unsigned lds = (unsigned)(size_t)&dst[r * BLD + c8];
            const __bf16* gp = memBf + (size_t)gr * EMB + kc + c8;
            asm volatile("global_load_async_to_lds_b128 %0, %1, off"
                         :: "v"(lds), "v"(gp) : "memory");
        }
    };

    float tv[TOPK]; int ti[TOPK];
#pragma unroll
    for (int t = 0; t < TOPK; ++t) { tv[t] = NEG_INF; ti[t] = 0; }

    v8f acc[2] = {};
    const __bf16* aBase = &smemQ[(mg * 16 + q16) * QLD];

    issue_fill(0);
    for (int g = 0; g < total; ++g) {
        if (g + 1 < total) {
            issue_fill(g + 1);
            asm volatile("s_wait_asynccnt 0x2" ::: "memory");  // fill(g) done
        } else {
            asm volatile("s_wait_asynccnt 0x0" ::: "memory");
        }
        __syncthreads();                                        // publish buf

        const int kc = (g & 15) * KCHUNK;
        const __bf16* bBuf = &smemB[g & 1][0];
#pragma unroll
        for (int ks = 0; ks < 2; ++ks) {                        // 2 k-steps
            const int kk = ks * 32 + half * 8;
            v16bf a = ld_op16(aBase + kc + kk);                 // reused 2x
#pragma unroll
            for (int j = 0; j < 2; ++j) {
                const int brow = ng * 32 + j * 16 + q16;
                v16bf b = ld_op16(bBuf + brow * BLD + kk);
                acc[j] = __builtin_amdgcn_wmma_f32_16x16x32_bf16(
                    false, a, false, b, (short)0, acc[j], false, false);
            }
        }

        if ((g & 15) == 15) {            // end of N tile: scores -> top-k
            const int n0 = splitBase + (g >> 4) * BN;
            const int rbase = half * 8;
#pragma unroll
            for (int j = 0; j < 2; ++j) {
#pragma unroll
                for (int r = 0; r < 8; ++r)
                    smemS[wave][(rbase + r) * 16 + q16] = acc[j][r];
                if (lane < 16) {           // wave-synchronous LDS readback
#pragma unroll
                    for (int c = 0; c < 16; ++c) {
                        int id = n0 + ng * 32 + j * 16 + c;
                        float v = (id < limit) ? smemS[wave][lane * 16 + c]
                                               : NEG_INF;
                        topk_insert(tv, ti, v, id);
                    }
                }
#pragma unroll
                for (int r = 0; r < 8; ++r) acc[j][r] = 0.f;
            }
        }
        __syncthreads();                 // protect buf[(g+2)&1] refill
    }

    // ---- merge the 2 N-group lists per row (alias dead B stage) ----
    float* mV = (float*)&smemB[0][0];                 // [2][BM][TOPK]
    int*   mI = (int*)((char*)&smemB[0][0] + 2 * BM * TOPK * sizeof(float));
    if (lane < 16) {
        const int row = mg * 16 + lane;
#pragma unroll
        for (int t = 0; t < TOPK; ++t) {
            mV[(ng * BM + row) * TOPK + t] = tv[t];
            mI[(ng * BM + row) * TOPK + t] = ti[t];
        }
    }
    __syncthreads();

    if (tid < BM) {
        float bv[TOPK]; int bi[TOPK];
#pragma unroll
        for (int t = 0; t < TOPK; ++t) { bv[t] = NEG_INF; bi[t] = 0; }
        for (int g2 = 0; g2 < 2; ++g2)
#pragma unroll
            for (int t = 0; t < TOPK; ++t)
                topk_insert(bv, bi, mV[(g2 * BM + tid) * TOPK + t],
                                    mI[(g2 * BM + tid) * TOPK + t]);
        const size_t base = ((size_t)(mBase + tid) * NSPLIT + split) * TOPK;
#pragma unroll
        for (int t = 0; t < TOPK; ++t) { pVal[base + t] = bv[t]; pIdx[base + t] = bi[t]; }
    }
}

__global__ void topk_merge_kernel(const float* __restrict__ pVal,
                                  const int* __restrict__ pIdx,
                                  int* __restrict__ idxOut) {
    int b = blockIdx.x * blockDim.x + threadIdx.x;
    if (b >= BATCH) return;
    float tv[TOPK]; int ti[TOPK];
#pragma unroll
    for (int t = 0; t < TOPK; ++t) { tv[t] = NEG_INF; ti[t] = 0; }
    const size_t base = (size_t)b * NSPLIT * TOPK;
    for (int c = 0; c < NSPLIT * TOPK; ++c)
        topk_insert(tv, ti, pVal[base + c], pIdx[base + c]);
#pragma unroll
    for (int t = 0; t < TOPK; ++t) idxOut[b * TOPK + t] = ti[t];
}

// ---------- generic WMMA projection GEMM: out = A[rowIdx] @ W^T + bias ----------
// One wave per 16x64 output strip (A fragment reused across 4 WMMAs).

__global__ __launch_bounds__(32)
void wmma_proj_gemm(const __bf16* __restrict__ A,
                    const int* __restrict__ rowIdx,
                    const __bf16* __restrict__ W,
                    const float* __restrict__ bias,
                    __bf16* __restrict__ outBf,
                    float* __restrict__ outF32) {
    const int lane = threadIdx.x;
    const int half = lane >> 4;
    const int q16  = lane & 15;

    const int am = blockIdx.x * 16 + q16;
    const int arow = rowIdx ? rowIdx[am] : am;
    const int ncol0 = blockIdx.y * 64;

    const __bf16* ab = A + (size_t)arow * EMB;
    const __bf16* wb[4];
#pragma unroll
    for (int j = 0; j < 4; ++j)
        wb[j] = W + (size_t)(ncol0 + j * 16 + q16) * EMB;

    v8f acc[4] = {};
#pragma unroll 2
    for (int k0 = 0; k0 < EMB; k0 += 32) {
        const int kk = k0 + half * 8;
        v16bf a = ld_op16(ab + kk);                      // reused 4x
#pragma unroll
        for (int j = 0; j < 4; ++j) {
            __builtin_prefetch(wb[j] + kk + 256, 0, 1);
            v16bf b = ld_op16(wb[j] + kk);
            acc[j] = __builtin_amdgcn_wmma_f32_16x16x32_bf16(
                false, a, false, b, (short)0, acc[j], false, false);
        }
    }

#pragma unroll
    for (int j = 0; j < 4; ++j) {
        const int col = ncol0 + j * 16 + q16;
        const float bc = bias ? bias[col] : 0.f;
#pragma unroll
        for (int r = 0; r < 8; ++r) {
            const int row = blockIdx.x * 16 + half * 8 + r;
            const float v = acc[j][r] + bc;
            if (outBf) outBf[(size_t)row * EMB + col] = f2bf(v);
            else       outF32[(size_t)row * EMB + col] = v;
        }
    }
}

// ---------- tiny attention: 1 query token vs TOPK keys, per (b, h) wave ----------

__global__ __launch_bounds__(256)
void attn_kernel(const __bf16* __restrict__ qProj,
                 const __bf16* __restrict__ kProj,
                 const __bf16* __restrict__ vProj,
                 __bf16* __restrict__ ctx) {
    const int b = blockIdx.x;
    const int h = threadIdx.x >> 5;          // 8 waves == 8 heads
    const int lane = threadIdx.x & 31;
    const int d0 = lane * 4;                 // 128 dims / 32 lanes
    const float scale = 0.088388347648318447f;   // 1/sqrt(128)

    const __bf16* qp = qProj + (size_t)b * EMB + h * HEAD_DIM + d0;
    float qv[4];
#pragma unroll
    for (int i = 0; i < 4; ++i) qv[i] = bf2f(qp[i]);

    float sc[TOPK];
#pragma unroll
    for (int j = 0; j < TOPK; ++j) {
        const __bf16* kp = kProj + ((size_t)(b * TOPK + j)) * EMB + h * HEAD_DIM + d0;
        float s = 0.f;
#pragma unroll
        for (int i = 0; i < 4; ++i) s += qv[i] * bf2f(kp[i]);
#pragma unroll
        for (int m = 16; m >= 1; m >>= 1) s += __shfl_xor(s, m, 32);
        sc[j] = s * scale;
    }

    float mx = sc[0];
#pragma unroll
    for (int j = 1; j < TOPK; ++j) mx = fmaxf(mx, sc[j]);
    float sum = 0.f;
#pragma unroll
    for (int j = 0; j < TOPK; ++j) { sc[j] = __expf(sc[j] - mx); sum += sc[j]; }
    const float inv = 1.f / sum;

    float cv[4] = {0.f, 0.f, 0.f, 0.f};
#pragma unroll
    for (int j = 0; j < TOPK; ++j) {
        const float a = sc[j] * inv;
        const __bf16* vp = vProj + ((size_t)(b * TOPK + j)) * EMB + h * HEAD_DIM + d0;
#pragma unroll
        for (int i = 0; i < 4; ++i) cv[i] += a * bf2f(vp[i]);
    }
    __bf16* cp = ctx + (size_t)b * EMB + h * HEAD_DIM + d0;
#pragma unroll
    for (int i = 0; i < 4; ++i) cp[i] = f2bf(cv[i]);
}

// ---------- host side ----------

extern "C" void kernel_launch(void* const* d_in, const int* in_sizes, int n_in,
                              void* d_out, int out_size, void* d_ws, size_t ws_size,
                              hipStream_t stream) {
    const float* query  = (const float*)d_in[0];
    const float* memory = (const float*)d_in[1];
    const float* w_q    = (const float*)d_in[2];
    const float* w_k    = (const float*)d_in[3];
    const float* w_v    = (const float*)d_in[4];
    const float* b_q    = (const float*)d_in[5];
    const float* b_k    = (const float*)d_in[6];
    const float* b_v    = (const float*)d_in[7];
    const float* w_o    = (const float*)d_in[8];
    const float* b_o    = (const float*)d_in[9];
    float* out = (float*)d_out;

    size_t off = 0;
    auto alloc = [&](size_t bytes) -> void* {
        void* p = (char*)d_ws + off;
        off += (bytes + 255) & ~(size_t)255;
        return p;
    };
    __bf16* memBf = (__bf16*)alloc((size_t)MEM_SIZE * EMB * 2);
    __bf16* qBf   = (__bf16*)alloc((size_t)BATCH * EMB * 2);
    __bf16* wqBf  = (__bf16*)alloc((size_t)EMB * EMB * 2);
    __bf16* wkBf  = (__bf16*)alloc((size_t)EMB * EMB * 2);
    __bf16* wvBf  = (__bf16*)alloc((size_t)EMB * EMB * 2);
    __bf16* woBf  = (__bf16*)alloc((size_t)EMB * EMB * 2);
    __bf16* qProj = (__bf16*)alloc((size_t)BATCH * EMB * 2);
    __bf16* kProj = (__bf16*)alloc((size_t)BATCH * TOPK * EMB * 2);
    __bf16* vProj = (__bf16*)alloc((size_t)BATCH * TOPK * EMB * 2);
    __bf16* ctx   = (__bf16*)alloc((size_t)BATCH * EMB * 2);
    float*  pVal  = (float*)alloc((size_t)BATCH * NSPLIT * TOPK * 4);
    int*    pIdx  = (int*)  alloc((size_t)BATCH * NSPLIT * TOPK * 4);
    int*    idx   = (int*)  alloc((size_t)BATCH * TOPK * 4);

    // 1) fp32 -> bf16 conversions
    cvt_f32_bf16<<<4096, 256, 0, stream>>>(memory, memBf, (long)MEM_SIZE * EMB);
    cvt_f32_bf16<<<512,  256, 0, stream>>>(query,  qBf,   (long)BATCH * EMB);
    cvt_f32_bf16<<<512,  256, 0, stream>>>(w_q,    wqBf,  (long)EMB * EMB);
    cvt_f32_bf16<<<512,  256, 0, stream>>>(w_k,    wkBf,  (long)EMB * EMB);
    cvt_f32_bf16<<<512,  256, 0, stream>>>(w_v,    wvBf,  (long)EMB * EMB);
    cvt_f32_bf16<<<512,  256, 0, stream>>>(w_o,    woBf,  (long)EMB * EMB);

    // 2) fused similarity GEMM + streaming top-k (dominant: 0.42 TFLOP)
    sim_topk_kernel<<<dim3(BATCH / BM, NSPLIT), 256, 0, stream>>>(qBf, memBf, pVal, pIdx);
    topk_merge_kernel<<<(BATCH + 255) / 256, 256, 0, stream>>>(pVal, pIdx, idx);

    // 3) projections (q; k/v with fused gather through idx)
    wmma_proj_gemm<<<dim3(BATCH / 16, EMB / 64), 32, 0, stream>>>(
        qBf, nullptr, wqBf, b_q, qProj, nullptr);
    wmma_proj_gemm<<<dim3(BATCH * TOPK / 16, EMB / 64), 32, 0, stream>>>(
        memBf, idx, wkBf, b_k, kProj, nullptr);
    wmma_proj_gemm<<<dim3(BATCH * TOPK / 16, EMB / 64), 32, 0, stream>>>(
        memBf, idx, wvBf, b_v, vProj, nullptr);

    // 4) attention over the 10 retrieved memories
    attn_kernel<<<BATCH, 256, 0, stream>>>(qProj, kProj, vProj, ctx);

    // 5) output projection -> fp32 d_out
    wmma_proj_gemm<<<dim3(BATCH / 16, EMB / 64), 32, 0, stream>>>(
        ctx, nullptr, woBf, b_o, nullptr, out);

    (void)in_sizes; (void)n_in; (void)out_size; (void)ws_size;
}
